// CrossViewFusion_22153441313269
// MI455X (gfx1250) — matrix operand order repaired and checked by reading, not compiled, not run on hardware
//
#include <hip/hip_runtime.h>
#include <math.h>

// MI455X / gfx1250: wave32, WMMA 16x16x32 bf16, 320KB LDS per WGP.
typedef __attribute__((ext_vector_type(16))) __bf16 v16bf;
typedef __attribute__((ext_vector_type(8)))  float  v8f;

#define BATCH    65536
#define DDIM     512
#define TB       64       // batch items per workgroup
#define ROWSR    128      // seq rows per workgroup
#define NTHREADS 256      // 8 waves
#define NGROUP   4        // head groups (2 heads each)

// LDS layout (bytes), total = 320KB (WGP max)
#define XBF_OFF   0        // xbf  [128][512] bf16 = 128KB ; becomes h in place
#define QKVC_OFF  131072   // kv   [128][256] bf16 =  64KB ; per head-group k|v
#define AOBF_OFF  196608   // ao   [128][512] bf16 = 128KB ; q written here, attn in-place
                           //   also: f32 staging for async x loads; gcat [64][1024] later
#define LDS_BYTES 327680

// bf16 weight scratch layout in d_ws (elements)
#define WS_OUT_OFF  786432     // 1536*512
#define WS_GATE_OFF 1048576    // + 512*512
#define WS_ELEMS    1572864    // + 512*1024

// ---- WMMA operand builders (wave32; contiguous k order, same in A and B) ----

// A-matrix 16x32 bf16 from a bf16 row in LDS.
// lanes 0-15:  halves0-15 = K[kb+0..15]
// lanes 16-31: halves0-15 = K[kb+16..31]
__device__ __forceinline__ v16bf make_a(const __bf16* row, int kb, int lane) {
    const int koff = (lane < 16) ? 0 : 16;
    v16bf a;
    ((uint4*)&a)[0] = *(const uint4*)(row + kb + koff);
    ((uint4*)&a)[1] = *(const uint4*)(row + kb + koff + 8);
    return a;
}

// B-matrix 32x16 bf16 (KxN) from weight row W[n][k].
// lanes 0-15: N=lane, K=kb+0..15 ; lanes 16-31: N=lane-16, K=kb+16..31
__device__ __forceinline__ v16bf make_b(const __bf16* wrow, int kb, int lane) {
    const int koff = (lane < 16) ? 0 : 16;
    v16bf b;
    ((uint4*)&b)[0] = *(const uint4*)(wrow + kb + koff);
    ((uint4*)&b)[1] = *(const uint4*)(wrow + kb + koff + 8);
    return b;
}
__device__ __forceinline__ v16bf make_b(const float* wrow, int kb, int lane) {
    const int koff = (lane < 16) ? 0 : 16;
    float tmp[16];
    *(float4*)(tmp + 0)  = *(const float4*)(wrow + kb + koff + 0);
    *(float4*)(tmp + 4)  = *(const float4*)(wrow + kb + koff + 4);
    *(float4*)(tmp + 8)  = *(const float4*)(wrow + kb + koff + 8);
    *(float4*)(tmp + 12) = *(const float4*)(wrow + kb + koff + 12);
    v16bf b;
#pragma unroll
    for (int i = 0; i < 16; ++i) b[i] = (__bf16)tmp[i];
    return b;
}

__device__ __forceinline__ v8f wmma_bf16(v16bf a, v16bf b, v8f c) {
    return __builtin_amdgcn_wmma_f32_16x16x32_bf16(false, a, false, b, (short)0, c,
                                                   false, false);
}

// ---- prologue: convert f32 weights -> bf16 scratch (L2-resident) ----
__global__ __launch_bounds__(256) void weights_to_bf16(
    const float* __restrict__ w_in, const float* __restrict__ w_out,
    const float* __restrict__ w_gate, __bf16* __restrict__ ws)
{
    const int total4 = WS_ELEMS / 4;
    for (int idx = blockIdx.x * 256 + threadIdx.x; idx < total4;
         idx += gridDim.x * 256) {
        const int e = idx * 4;
        const float* src;
        int off;
        if (e < WS_OUT_OFF)       { src = w_in;   off = e; }
        else if (e < WS_GATE_OFF) { src = w_out;  off = e - WS_OUT_OFF; }
        else                      { src = w_gate; off = e - WS_GATE_OFF; }
        float4 v = *(const float4*)(src + off);
        union { __bf16 h[4]; uint2 u; } cv;
        cv.h[0] = (__bf16)v.x; cv.h[1] = (__bf16)v.y;
        cv.h[2] = (__bf16)v.z; cv.h[3] = (__bf16)v.w;
        *(uint2*)(ws + e) = cv.u;
    }
}

// ---- main fused kernel, templated on weight element type ----
template <typename WT>
__global__ __launch_bounds__(NTHREADS) void cvf_kernel(
    const float* __restrict__ x_cc,  const float* __restrict__ x_mlo,
    const float* __restrict__ vmask,
    const WT* __restrict__ w_in,   const float* __restrict__ b_in,
    const WT* __restrict__ w_out,  const float* __restrict__ b_out,
    const float* __restrict__ ln_g,  const float* __restrict__ ln_b,
    const WT* __restrict__ w_gate, const float* __restrict__ b_gate,
    float* __restrict__ out)
{
    extern __shared__ unsigned char smem[];
    __bf16* xbf  = (__bf16*)(smem + XBF_OFF);    // [128][512]
    __bf16* qkvc = (__bf16*)(smem + QKVC_OFF);   // [128][256] k|v chunk
    __bf16* aobf = (__bf16*)(smem + AOBF_OFF);   // [128][512] q -> ao
    __bf16* gcat = (__bf16*)(smem + AOBF_OFF);   // [64][1024] gate input (reuse)

    const int tid  = threadIdx.x;
    const int lane = tid & 31;
    const int wave = tid >> 5;
    const int b0   = blockIdx.x * TB;

    // ------- Phase A: async-stage x (f32) into LDS, convert to bf16 --------
    {
        const unsigned stage_lds = (unsigned)(uintptr_t)(smem + AOBF_OFF);
        const float* stf = (const float*)(smem + AOBF_OFF);
        for (int half = 0; half < 2; ++half) {
            // 64 rows x 512 f32 = 128KB = 8192 x 16B chunks
#pragma unroll 4
            for (int it = 0; it < 32; ++it) {
                int c16 = tid + it * NTHREADS;       // chunk id 0..8191
                int r  = c16 >> 7;                   // 128 chunks per row
                int cf = (c16 & 127) << 2;           // float col
                int gr = half * 64 + r;
                int b  = b0 + (gr >> 1);
                const float* src = (gr & 1) ? x_mlo : x_cc;
                unsigned long long gaddr =
                    (unsigned long long)(uintptr_t)(src + (size_t)b * DDIM + cf);
                unsigned laddr = stage_lds + (unsigned)c16 * 16u;
                asm volatile("global_load_async_to_lds_b128 %0, %1, off"
                             :: "v"(laddr), "v"(gaddr) : "memory");
            }
            asm volatile("s_wait_asynccnt 0" ::: "memory");
            __syncthreads();
            // convert staged f32 -> xbf (bf16)
#pragma unroll 4
            for (int it = 0; it < 32; ++it) {
                int i4 = tid + it * NTHREADS;        // float4 id 0..8191
                int r  = i4 >> 7;
                int cf = (i4 & 127) << 2;
                float4 v = ((const float4*)stf)[i4];
                union { __bf16 h[4]; uint2 u; } cv;
                cv.h[0] = (__bf16)v.x; cv.h[1] = (__bf16)v.y;
                cv.h[2] = (__bf16)v.z; cv.h[3] = (__bf16)v.w;
                *(uint2*)(xbf + (half * 64 + r) * DDIM + cf) = cv.u;
            }
            __syncthreads();   // staging reusable / xbf visible
        }
    }

    // ------- Phases B+C per head-group: qkv GEMM slice + attention ---------
    for (int g = 0; g < NGROUP; ++g) {
        // GEMM1 slice: 24 N-tiles (8 q, 8 k, 8 v), 3 per wave; 8 M-tiles each,
        // one B load feeds 8 WMMAs.
#pragma unroll
        for (int q3 = 0; q3 < 3; ++q3) {
            const int nt  = wave * 3 + q3;
            const int sec = nt >> 3;                 // 0=q 1=k 2=v
            const int sub = nt & 7;
            const int ncl = sub * 16 + (lane & 15);  // col within 128-wide slice
            const int ng  = sec * 512 + g * 128 + ncl;  // global weight row
            const WT* wrow = w_in + (size_t)ng * DDIM;
            v8f acc[8] = {};
            for (int ks = 0; ks < 16; ++ks) {
                const int kb = ks * 32;
                __builtin_prefetch(wrow + kb + 64, 0, 1);
                v16bf bm = make_b(wrow, kb, lane);
#pragma unroll
                for (int mt = 0; mt < 8; ++mt) {
                    const __bf16* arow = xbf + (mt * 16 + (lane & 15)) * DDIM;
                    acc[mt] = wmma_bf16(make_a(arow, kb, lane), bm, acc[mt]);
                }
            }
            const float bias = b_in[ng];
            __bf16* dst;
            int stride;
            if (sec == 0)      { dst = aobf + g * 128 + ncl;       stride = 512; }
            else if (sec == 1) { dst = qkvc + ncl;                 stride = 256; }
            else               { dst = qkvc + 128 + ncl;           stride = 256; }
            const int mo = (lane < 16) ? 0 : 8;
#pragma unroll
            for (int mt = 0; mt < 8; ++mt)
#pragma unroll
                for (int j = 0; j < 8; ++j)
                    dst[(mt * 16 + mo + j) * stride] = (__bf16)(acc[mt][j] + bias);
        }
        __syncthreads();

        // attention for this group's 2 heads (seq len 2)
        {
            const int bl = tid >> 2;          // 0..63
            const int hg = (tid >> 1) & 1;    // head within group
            const int vq = tid & 1;
            const int rq = bl * 2 + vq;
            const int hc = hg * 64;
            const __bf16* qp = aobf + rq * 512 + g * 128 + hc;
            const __bf16* k0 = qkvc + (bl * 2) * 256 + hc;
            const __bf16* k1 = qkvc + (bl * 2 + 1) * 256 + hc;
            float s0 = 0.f, s1 = 0.f;
#pragma unroll 8
            for (int i = 0; i < 64; ++i) {
                float q = (float)qp[i];
                s0 += q * (float)k0[i];
                s1 += q * (float)k1[i];
            }
            s0 *= 0.125f; s1 *= 0.125f;
            const float vm0 = vmask[(size_t)(b0 + bl) * 2 + 0];
            const float vm1 = vmask[(size_t)(b0 + bl) * 2 + 1];
            if (vm0 == 0.f) s0 = -__builtin_inff();
            if (vm1 == 0.f) s1 = -__builtin_inff();
            const float mx = fmaxf(s0, s1);
            const float e0 = __expf(s0 - mx), e1 = __expf(s1 - mx);
            const float inv = 1.f / (e0 + e1);
            const float a0 = e0 * inv, a1 = e1 * inv;
            const __bf16* v0 = qkvc + (bl * 2) * 256 + 128 + hc;
            const __bf16* v1 = qkvc + (bl * 2 + 1) * 256 + 128 + hc;
            __bf16* aop = aobf + rq * 512 + g * 128 + hc;  // overwrite own q
#pragma unroll 8
            for (int i = 0; i < 64; ++i)
                aop[i] = (__bf16)(a0 * (float)v0[i] + a1 * (float)v1[i]);
        }
        __syncthreads();  // kv chunk reusable for next group
    }

    // ------- Phase D: h = x + AO @ Wout^T + b_out (into xbf, in place) -----
#pragma unroll
    for (int q4 = 0; q4 < 4; ++q4) {
        const int nt = wave * 4 + q4;               // 32 N-tiles
        const int n  = nt * 16 + (lane & 15);
        const WT* wrow = w_out + (size_t)n * DDIM;
        v8f acc[8] = {};
        for (int ks = 0; ks < 16; ++ks) {
            const int kb = ks * 32;
            __builtin_prefetch(wrow + kb + 64, 0, 1);
            v16bf bm = make_b(wrow, kb, lane);
#pragma unroll
            for (int mt = 0; mt < 8; ++mt) {
                const __bf16* arow = aobf + (mt * 16 + (lane & 15)) * 512;
                acc[mt] = wmma_bf16(make_a(arow, kb, lane), bm, acc[mt]);
            }
        }
        const float bias = b_out[n];
        const int mo = (lane < 16) ? 0 : 8;
        const int nc = nt * 16 + (lane & 15);
#pragma unroll
        for (int mt = 0; mt < 8; ++mt)
#pragma unroll
            for (int j = 0; j < 8; ++j) {
                const int idx = (mt * 16 + mo + j) * DDIM + nc;
                xbf[idx] = (__bf16)((float)xbf[idx] + acc[mt][j] + bias);
            }
    }
    __syncthreads();

    // ------- Phase E: LayerNorm rows of h; emit gate input -----------------
    {
        const int r  = tid >> 1;             // 128 rows, 2 threads per row
        const int c0 = (tid & 1) * 256;
        const __bf16* hr = xbf + r * DDIM;
        float s = 0.f, sq = 0.f;
#pragma unroll 8
        for (int i = 0; i < 256; ++i) {
            float v = (float)hr[c0 + i];
            s += v; sq += v * v;
        }
        s  += __shfl_xor(s, 1, 2);
        sq += __shfl_xor(sq, 1, 2);
        const float mu  = s * (1.f / 512.f);
        const float var = sq * (1.f / 512.f) - mu * mu;
        const float rs  = rsqrtf(var + 1e-5f);
        __bf16* hw = xbf + r * DDIM;
        __bf16* gr = gcat + (r >> 1) * 1024 + (r & 1) * 512;
#pragma unroll 4
        for (int i = 0; i < 256; ++i) {
            const int d = c0 + i;
            float v = ((float)hr[d] - mu) * rs * ln_g[d] + ln_b[d];
            hw[d] = (__bf16)v;
            gr[d] = (__bf16)v;
        }
    }
    __syncthreads();

    // ------- Phase F: gate GEMM (M=64,N=512,K=1024) + fuse + output --------
#pragma unroll
    for (int q4 = 0; q4 < 4; ++q4) {
        const int nt = wave * 4 + q4;
        const int n  = nt * 16 + (lane & 15);
        const WT* wrow = w_gate + (size_t)n * 1024;
        v8f acc[4] = {};
        for (int ks = 0; ks < 32; ++ks) {
            const int kb = ks * 32;
            __builtin_prefetch(wrow + kb + 64, 0, 1);
            v16bf bm = make_b(wrow, kb, lane);
#pragma unroll
            for (int mt = 0; mt < 4; ++mt) {
                const __bf16* arow = gcat + (mt * 16 + (lane & 15)) * 1024;
                acc[mt] = wmma_bf16(make_a(arow, kb, lane), bm, acc[mt]);
            }
        }
        const float bias = b_gate[n];
        const int mo = (lane < 16) ? 0 : 8;
        const int nc = nt * 16 + (lane & 15);
#pragma unroll
        for (int mt = 0; mt < 4; ++mt)
#pragma unroll
            for (int j = 0; j < 8; ++j) {
                const int bl = mt * 16 + mo + j;          // batch-local item
                const float gv = 1.f / (1.f + __expf(-(acc[mt][j] + bias)));
                const float cc = (float)xbf[(bl * 2) * DDIM + nc];
                const float ml = (float)xbf[(bl * 2 + 1) * DDIM + nc];
                const float vm0 = vmask[(size_t)(b0 + bl) * 2 + 0];
                const float vm1 = vmask[(size_t)(b0 + bl) * 2 + 1];
                const float fused = gv * cc + (1.f - gv) * ml;
                const float res = (vm0 * vm1 > 0.5f) ? fused
                                                     : (cc * vm0 + ml * vm1);
                out[(size_t)(b0 + bl) * DDIM + nc] = res;
            }
    }
}

extern "C" void kernel_launch(void* const* d_in, const int* in_sizes, int n_in,
                              void* d_out, int out_size, void* d_ws, size_t ws_size,
                              hipStream_t stream) {
    const float* x_cc  = (const float*)d_in[0];
    const float* x_mlo = (const float*)d_in[1];
    const float* vm    = (const float*)d_in[2];
    const float* w_in  = (const float*)d_in[3];
    const float* b_in  = (const float*)d_in[4];
    const float* w_out = (const float*)d_in[5];
    const float* b_out = (const float*)d_in[6];
    const float* ln_g  = (const float*)d_in[7];
    const float* ln_b  = (const float*)d_in[8];
    const float* w_g   = (const float*)d_in[9];
    const float* b_g   = (const float*)d_in[10];
    float* out = (float*)d_out;

    dim3 grid(BATCH / TB);
    dim3 block(NTHREADS);

    if (ws_size >= (size_t)WS_ELEMS * sizeof(__bf16)) {
        __bf16* wsb = (__bf16*)d_ws;
        hipLaunchKernelGGL(weights_to_bf16, dim3(1536), dim3(256), 0, stream,
                           w_in, w_out, w_g, wsb);
        hipLaunchKernelGGL((cvf_kernel<__bf16>), grid, block, LDS_BYTES, stream,
                           x_cc, x_mlo, vm,
                           wsb, b_in, wsb + WS_OUT_OFF, b_out,
                           ln_g, ln_b, wsb + WS_GATE_OFF, b_g, out);
    } else {
        hipLaunchKernelGGL((cvf_kernel<float>), grid, block, LDS_BYTES, stream,
                           x_cc, x_mlo, vm,
                           w_in, b_in, w_out, b_out,
                           ln_g, ln_b, w_g, b_g, out);
    }
}
